// MolecularGraphNeuralNetwork_2860448219498
// MI455X (gfx1250) — matrix operand order repaired and checked by reading, not compiled, not run on hardware
//
#include <hip/hip_runtime.h>
#include <hip/hip_bf16.h>
#include <math.h>

#define LRELU_ALPHA 0.25f
#define NEGV  -9.0e10f
#define N_NODES 8192
#define DIM 128
#define N_MOL 256

typedef __bf16 bf16_t;
typedef __attribute__((ext_vector_type(16))) __bf16 v16bf;
typedef __attribute__((ext_vector_type(8)))  __bf16 v8bf;
typedef __attribute__((ext_vector_type(8)))  float  v8f;
typedef __attribute__((ext_vector_type(4)))  int    v4i;

// Address-space-qualified int4 pointers matching the async-load builtin's
// (int4 AS1*, int4 AS3*, imm, imm) signature discovered from the diagnostics.
typedef __attribute__((address_space(1))) v4i* gv4i_p;
typedef __attribute__((address_space(3))) v4i* sv4i_p;

#if defined(__has_builtin)
#if __has_builtin(__builtin_amdgcn_global_load_async_to_lds_b128)
#define USE_ASYNC_LDS 1
#endif
#endif

union AF { v16bf v; v8bf h[2]; };

static __device__ inline v8f vzero8() {
  v8f z; for (int i = 0; i < 8; i++) z[i] = 0.f; return z;
}

// A-fragment (16x32 bf16, M x K) from row-major global memory.
// ISA layout: lane&15 = row; lanes>=16 take K halves 8..15 / 24..31.
// Each lane: elems 0..7 = A[r][kh..kh+7], elems 8..15 = A[r][kh+16..kh+23].
static __device__ inline v16bf load_a_global(const bf16_t* base, int stride, int lane) {
  int r  = lane & 15;
  int kh = (lane >> 4) << 3;
  const bf16_t* p = base + (size_t)r * stride + kh;
  AF f;
  f.h[0] = *(const v8bf*)(p);
  f.h[1] = *(const v8bf*)(p + 16);
  return f.v;
}

// B-fragment (32x16 bf16, K x N) from LDS holding B transposed: Bt[n][k].
// ISA layout: lane&15 = column; lanes>=16 take K 8..15 / 24..31 (same as A).
static __device__ inline v16bf load_b_lds(const bf16_t* Bt, int stride, int k0, int lane) {
  int n  = lane & 15;
  int kb = k0 + ((lane >> 4) << 3);
  const bf16_t* p = Bt + n * stride + kb;
  AF f;
  f.h[0] = *(const v8bf*)(p);
  f.h[1] = *(const v8bf*)(p + 16);
  return f.v;
}

// ---------------------------------------------------------------------------
// Dense layer: C[M x N] = act(A[M x 128]_bf16 @ W[128 x ncols]_f32 + b)
// W staged into LDS as bf16 transposed (zero-padded to N columns).
// ---------------------------------------------------------------------------
template<int NT, int ACT, int BIAS>
__global__ void __launch_bounds__(128) k_dense(const bf16_t* __restrict__ A,
                                               const float* __restrict__ W,
                                               const float* __restrict__ bias,
                                               float* __restrict__ C, int ncols) {
  constexpr int N  = NT * 16;
  constexpr int KP = DIM + 8;            // padded LDS row, 16B-aligned rows
  __shared__ bf16_t Wt[N * KP];
  __shared__ float  bsm[N];
  int tid = threadIdx.x;
  for (int idx = tid; idx < DIM * N; idx += 128) {
    int k = idx / N, n = idx % N;
    Wt[n * KP + k] = (n < ncols) ? (bf16_t)W[k * ncols + n] : (bf16_t)0.f;
  }
  if (tid < N) bsm[tid] = (BIAS && tid < ncols) ? bias[tid] : 0.f;
  __syncthreads();

  int wave = tid >> 5, lane = tid & 31;
  int m0 = blockIdx.x * 64 + wave * 16;
  v8f acc[NT];
  for (int t = 0; t < NT; t++) acc[t] = vzero8();
  const bf16_t* Arow = A + (size_t)m0 * DIM;
  for (int k0 = 0; k0 < DIM; k0 += 32) {
    v16bf a = load_a_global(Arow + k0, DIM, lane);
#pragma unroll
    for (int t = 0; t < NT; t++) {
      v16bf b = load_b_lds(&Wt[t * 16 * KP], KP, k0, lane);
      acc[t] = __builtin_amdgcn_wmma_f32_16x16x32_bf16(
          false, a, false, b, (short)0, acc[t], false, false);
    }
  }
  int n_l = lane & 15, r0 = (lane >> 4) * 8;
  for (int t = 0; t < NT; t++)
    for (int r = 0; r < 8; r++) {
      float v = acc[t][r] + bsm[t * 16 + n_l];
      if (ACT == 1) v = fmaxf(v, 0.f);
      C[(size_t)(m0 + r0 + r) * N + t * 16 + n_l] = v;
    }
}

// ---------------------------------------------------------------------------
// Big N x N GEMM over the graph. MODE 0: A = adj (bf16, L2-resident).
// MODE 1: A = softmax attention, recomputed on the fly from (s1,s2,adj,rmax,rinv)
// and packed into bf16 WMMA A-fragments in registers (never materialized).
// B is supplied TRANSPOSED in global memory: Bt_g[n][k], n < N, k < 8192, so
// the LDS tile is staged with verbatim 16B copies -> async global->LDS path.
// EPI 0: C = acc + R (residual add);  EPI 1: C += elu(acc);  EPI 2: C = elu(acc)
// ---------------------------------------------------------------------------
template<int NT, int MODE, int EPI>
__global__ void __launch_bounds__(128) k_nn_gemm(const bf16_t* __restrict__ Aadj,
                                                 const bf16_t* __restrict__ Bt_g,
                                                 const float* __restrict__ s1,
                                                 const float* __restrict__ s2,
                                                 const float* __restrict__ rmax,
                                                 const float* __restrict__ rinv,
                                                 const float* __restrict__ R,
                                                 float* __restrict__ C) {
  constexpr int N   = NT * 16;
  constexpr int KC  = 128;
  constexpr int KCP = KC + 8;
  __shared__ bf16_t Bt[N * KCP];
  __shared__ float  s2sm[KC];
  int tid = threadIdx.x;
  int wave = tid >> 5, lane = tid & 31;
  int m0 = blockIdx.x * 64 + wave * 16;
  int mrow = m0 + (lane & 15);
  int khalf = (lane >> 4) << 3;
  float s1m = 0.f, mxm = 0.f, rim = 0.f;
  if (MODE == 1) { s1m = s1[mrow]; mxm = rmax[mrow]; rim = rinv[mrow]; }
  v8f acc[NT];
  for (int t = 0; t < NT; t++) acc[t] = vzero8();
  const bf16_t* arow = Aadj + (size_t)mrow * N_NODES + khalf;

  for (int kc0 = 0; kc0 < N_NODES; kc0 += KC) {
    __syncthreads();
    // Stage Bt chunk: row n of LDS <- Bt_g[n][kc0 .. kc0+KC), 16B granules.
#if defined(USE_ASYNC_LDS)
    for (int c = tid; c < N * (KC / 8); c += 128) {
      int n = c / (KC / 8), kl = (c % (KC / 8)) * 8;
      const bf16_t* g = Bt_g + (size_t)n * N_NODES + kc0 + kl;
      bf16_t* l = &Bt[n * KCP + kl];
      __builtin_amdgcn_global_load_async_to_lds_b128((gv4i_p)g, (sv4i_p)l, 0, 0);
    }
    asm volatile("s_wait_asynccnt 0x0" ::: "memory");
#else
    for (int c = tid; c < N * (KC / 8); c += 128) {
      int n = c / (KC / 8), kl = (c % (KC / 8)) * 8;
      *(v8bf*)(&Bt[n * KCP + kl]) =
          *(const v8bf*)(Bt_g + (size_t)n * N_NODES + kc0 + kl);
    }
#endif
    if (MODE == 1)
      for (int idx = tid; idx < KC; idx += 128) s2sm[idx] = s2[kc0 + idx];
    __syncthreads();
    __builtin_prefetch(arow + kc0 + KC, 0, 0);   // next adj chunk (L2 hit)
#pragma unroll 1
    for (int k0 = 0; k0 < KC; k0 += 32) {
      const bf16_t* ap = arow + kc0 + k0;
      AF adjf;
      adjf.h[0] = *(const v8bf*)(ap);
      adjf.h[1] = *(const v8bf*)(ap + 16);
      v16bf afrag;
      if (MODE == 1) {
        AF f;
#pragma unroll
        for (int h = 0; h < 2; h++)
#pragma unroll
          for (int j = 0; j < 8; j++) {
            float e = s1m + s2sm[k0 + khalf + h * 16 + j];
            e = (e > 0.f) ? e : LRELU_ALPHA * e;
            float wv = ((float)adjf.h[h][j] > 0.f) ? __expf(e - mxm) * rim : 0.f;
            f.h[h][j] = (bf16_t)wv;
          }
        afrag = f.v;
      } else {
        afrag = adjf.v;
      }
#pragma unroll
      for (int t = 0; t < NT; t++) {
        v16bf b = load_b_lds(&Bt[t * 16 * KCP], KCP, k0, lane);
        acc[t] = __builtin_amdgcn_wmma_f32_16x16x32_bf16(
            false, afrag, false, b, (short)0, acc[t], false, false);
      }
    }
  }
  int n_l = lane & 15, r0 = (lane >> 4) * 8;
  for (int t = 0; t < NT; t++)
    for (int r = 0; r < 8; r++) {
      size_t o = (size_t)(m0 + r0 + r) * N + t * 16 + n_l;
      float v = acc[t][r];
      if (EPI == 0)      C[o] = v + R[o];
      else if (EPI == 1) C[o] += (v > 0.f ? v : __expf(v) - 1.f);
      else               C[o] = (v > 0.f ? v : __expf(v) - 1.f);
    }
}

// ---------------------------------------------------------------------------
// Per-row softmax stats over masked leaky-relu(s1_i + s2_j): row max and 1/sum.
// ---------------------------------------------------------------------------
__global__ void __launch_bounds__(256) k_stats(const bf16_t* __restrict__ Aadj,
                                               const float* __restrict__ s1,
                                               const float* __restrict__ s2,
                                               float* __restrict__ rmax,
                                               float* __restrict__ rinv) {
  __shared__ float sm[256];
  int row = blockIdx.x, tid = threadIdx.x;
  const bf16_t* ar = Aadj + (size_t)row * N_NODES;
  float s1r = s1[row];
  float mx = -3.4e38f;
  for (int j = tid; j < N_NODES; j += 256) {
    float e = s1r + s2[j];
    e = (e > 0.f) ? e : LRELU_ALPHA * e;
    e = ((float)ar[j] > 0.f) ? e : NEGV;
    mx = fmaxf(mx, e);
  }
  sm[tid] = mx; __syncthreads();
  for (int s = 128; s > 0; s >>= 1) { if (tid < s) sm[tid] = fmaxf(sm[tid], sm[tid + s]); __syncthreads(); }
  mx = sm[0]; __syncthreads();
  float sum = 0.f;
  for (int j = tid; j < N_NODES; j += 256) {
    float e = s1r + s2[j];
    e = (e > 0.f) ? e : LRELU_ALPHA * e;
    e = ((float)ar[j] > 0.f) ? e : NEGV;
    sum += __expf(e - mx);
  }
  sm[tid] = sum; __syncthreads();
  for (int s = 128; s > 0; s >>= 1) { if (tid < s) sm[tid] += sm[tid + s]; __syncthreads(); }
  if (tid == 0) { rmax[row] = mx; rinv[row] = 1.f / fmaxf(sm[0], 1e-30f); }
}

__global__ void __launch_bounds__(128) k_s1s2(const float* __restrict__ H, int ld, int len,
                                              const float* __restrict__ a1,
                                              const float* __restrict__ a2,
                                              float* __restrict__ s1, float* __restrict__ s2) {
  __shared__ float p1[128], p2[128];
  int row = blockIdx.x, t = threadIdx.x;
  float v = (t < len) ? H[(size_t)row * ld + t] : 0.f;
  p1[t] = (t < len) ? v * a1[t] : 0.f;
  p2[t] = (t < len) ? v * a2[t] : 0.f;
  __syncthreads();
  for (int s = 64; s > 0; s >>= 1) { if (t < s) { p1[t] += p1[t + s]; p2[t] += p2[t + s]; } __syncthreads(); }
  if (t == 0) { s1[row] = p1[0]; s2[row] = p2[0]; }
}

__global__ void __launch_bounds__(128) k_l2norm(float* __restrict__ X) {
  __shared__ float sm[128];
  int row = blockIdx.x, t = threadIdx.x;
  float v = X[(size_t)row * DIM + t];
  sm[t] = v * v; __syncthreads();
  for (int s = 64; s > 0; s >>= 1) { if (t < s) sm[t] += sm[t + s]; __syncthreads(); }
  float d = fmaxf(sqrtf(sm[0]), 1e-12f);
  X[(size_t)row * DIM + t] = v / d;
}

__global__ void __launch_bounds__(64) k_softmax56(float* __restrict__ O) {
  __shared__ float sm[64];
  int row = blockIdx.x, t = threadIdx.x;
  float v = (t < 56) ? O[(size_t)row * 64 + t] : -3.4e38f;
  sm[t] = v; __syncthreads();
  for (int s = 32; s > 0; s >>= 1) { if (t < s) sm[t] = fmaxf(sm[t], sm[t + s]); __syncthreads(); }
  float mx = sm[0]; __syncthreads();
  float e = (t < 56) ? __expf(v - mx) : 0.f;
  sm[t] = e; __syncthreads();
  for (int s = 32; s > 0; s >>= 1) { if (t < s) sm[t] += sm[t + s]; __syncthreads(); }
  O[(size_t)row * 64 + t] = (t < 56) ? e / sm[0] : 0.f;
}

__global__ void k_segsum(const float* __restrict__ X56, const int* __restrict__ seg,
                         float* __restrict__ mol) {
  int idx = blockIdx.x * blockDim.x + threadIdx.x;
  if (idx >= N_NODES * 56) return;
  int i = idx / 56, c = idx % 56;
  atomicAdd(&mol[seg[i] * 56 + c], X56[(size_t)i * 64 + c]);
}

__global__ void __launch_bounds__(64) k_head(const float* __restrict__ mol,
                                             const float* __restrict__ W_out,
                                             const float* __restrict__ b_out,
                                             const float* __restrict__ W_prop,
                                             const float* __restrict__ b_prop,
                                             float* __restrict__ out) {
  __shared__ float v0[56], v1[56];
  int b = blockIdx.x, t = threadIdx.x;
  if (t < 56) v0[t] = mol[b * 56 + t];
  __syncthreads();
  if (t < 56) {
    float s = b_out[t];
    for (int j = 0; j < 56; j++) s += v0[j] * W_out[j * 56 + t];
    v1[t] = fmaxf(s, 0.f);
  }
  __syncthreads();
  if (t < 56) {
    float s = b_out[56 + t];
    for (int j = 0; j < 56; j++) s += v1[j] * W_out[3136 + j * 56 + t];
    v0[t] = fmaxf(s, 0.f);
  }
  __syncthreads();
  if (t < 2) {
    float s = b_prop[t];
    for (int j = 0; j < 56; j++) s += v0[j] * W_prop[j * 2 + t];
    out[b * 2 + t] = 1.f / (1.f + __expf(-s));
  }
}

__global__ void k_gather(const int* __restrict__ fp, const float* __restrict__ embed,
                         float* __restrict__ X) {
  int i = blockIdx.x, t = threadIdx.x;
  X[(size_t)i * DIM + t] = embed[(size_t)fp[i] * DIM + t];
}

__global__ void k_cvt_adj(const float* __restrict__ A, bf16_t* __restrict__ O, size_t n) {
  size_t i = ((size_t)blockIdx.x * blockDim.x + threadIdx.x) * 8;
  if (i >= n) return;
  v8bf o;
  for (int j = 0; j < 8; j++) o[j] = (bf16_t)A[i + j];
  *(v8bf*)(O + i) = o;
}

__global__ void k_cvt(const float* __restrict__ X, bf16_t* __restrict__ O,
                      float scale, size_t n) {
  size_t i = ((size_t)blockIdx.x * blockDim.x + threadIdx.x) * 8;
  if (i >= n) return;
  v8bf o;
  for (int j = 0; j < 8; j++) o[j] = (bf16_t)(scale * X[i + j]);
  *(v8bf*)(O + i) = o;
}

// Transposing convert: X [8192 x ld] f32  ->  T [ld x 8192] bf16 (both coalesced)
__global__ void k_cvt_t(const float* __restrict__ X, bf16_t* __restrict__ T,
                        int ld, float scale) {
  __shared__ bf16_t tile[32][33];
  int bx = blockIdx.x * 32;   // node base
  int by = blockIdx.y * 32;   // feature base
  int tx = threadIdx.x, ty = threadIdx.y;   // 32 x 8
  for (int i = ty; i < 32; i += 8)
    tile[i][tx] = (bf16_t)(scale * X[(size_t)(bx + i) * ld + by + tx]);
  __syncthreads();
  for (int i = ty; i < 32; i += 8)
    T[(size_t)(by + i) * N_NODES + bx + tx] = tile[tx][i];
}

extern "C" void kernel_launch(void* const* d_in, const int* in_sizes, int n_in,
                              void* d_out, int out_size, void* d_ws, size_t ws_size,
                              hipStream_t stream) {
  const int*   fp    = (const int*)d_in[0];
  const float* adj   = (const float*)d_in[1];
  const int*   seg   = (const int*)d_in[2];
  const float* embed = (const float*)d_in[3];
  const float* W_fp  = (const float*)d_in[4];
  const float* b_fp  = (const float*)d_in[5];
  const float* Wh    = (const float*)d_in[6];
  const float* ah    = (const float*)d_in[7];
  const float* W_oa  = (const float*)d_in[8];
  const float* a_oa  = (const float*)d_in[9];
  const float* W_out = (const float*)d_in[10];
  const float* b_out = (const float*)d_in[11];
  const float* W_pr  = (const float*)d_in[12];
  const float* b_pr  = (const float*)d_in[13];
  float* out = (float*)d_out;
  (void)in_sizes; (void)n_in; (void)out_size; (void)ws_size;

  char* w = (char*)d_ws; size_t off = 0;
  auto alloc = [&](size_t bytes) -> void* {
    void* p = w + off; off += (bytes + 255) & ~(size_t)255; return p;
  };
  bf16_t* A16 = (bf16_t*)alloc((size_t)N_NODES * N_NODES * 2);   // 128 MB, fits 192 MB L2
  float*  X   = (float*)alloc((size_t)N_NODES * DIM * 4);
  float*  H   = (float*)alloc((size_t)N_NODES * DIM * 4);
  float*  Z   = (float*)alloc((size_t)N_NODES * DIM * 4);
  bf16_t* Xb  = (bf16_t*)alloc((size_t)N_NODES * DIM * 2);   // row-major (GEMM A side)
  bf16_t* Hbt = (bf16_t*)alloc((size_t)N_NODES * DIM * 2);   // transposed (GEMM B side)
  float*  H56 = (float*)alloc((size_t)N_NODES * 64 * 4);
  float*  O56 = (float*)alloc((size_t)N_NODES * 64 * 4);
  float*  s1  = (float*)alloc(N_NODES * 4);
  float*  s2  = (float*)alloc(N_NODES * 4);
  float*  rmx = (float*)alloc(N_NODES * 4);
  float*  rin = (float*)alloc(N_NODES * 4);
  float*  mol = (float*)alloc(N_MOL * 56 * 4);

  const size_t nXD = (size_t)N_NODES * DIM;
  const int cvtBlocks = (int)(nXD / 8 / 256);
  dim3 tb(32, 8);

  // adjacency -> bf16 once (exact for 0/1); all later N^2 passes hit L2
  {
    size_t n = (size_t)N_NODES * N_NODES;
    k_cvt_adj<<<(unsigned)(n / 8 / 256), 256, 0, stream>>>(adj, A16, n);
  }
  // embedding gather
  k_gather<<<N_NODES, DIM, 0, stream>>>(fp, embed, X);

  // three message-passing layers: x = l2norm(relu(xW+b) + adj@relu(xW+b))
  for (int l = 0; l < 3; l++) {
    k_cvt<<<cvtBlocks, 256, 0, stream>>>(X, Xb, 1.f, nXD);
    k_dense<8, 1, 1><<<N_NODES / 64, 128, 0, stream>>>(Xb, W_fp + l * DIM * DIM,
                                                       b_fp + l * DIM, H, DIM);
    k_cvt_t<<<dim3(N_NODES / 32, DIM / 32), tb, 0, stream>>>(H, Hbt, DIM, 1.f);
    k_nn_gemm<8, 0, 0><<<N_NODES / 64, 128, 0, stream>>>(A16, Hbt, nullptr, nullptr,
                                                         nullptr, nullptr, H, X);
    k_l2norm<<<N_NODES, DIM, 0, stream>>>(X);
  }

  // GAT heads: Z += elu(att @ (x Wh)); x = Z/2
  (void)hipMemsetAsync(Z, 0, nXD * 4, stream);
  k_cvt<<<cvtBlocks, 256, 0, stream>>>(X, Xb, 1.f, nXD);
  for (int h = 0; h < 2; h++) {
    k_dense<8, 0, 0><<<N_NODES / 64, 128, 0, stream>>>(Xb, Wh + h * DIM * DIM,
                                                       nullptr, H, DIM);
    k_cvt_t<<<dim3(N_NODES / 32, DIM / 32), tb, 0, stream>>>(H, Hbt, DIM, 1.f);
    k_s1s2<<<N_NODES, 128, 0, stream>>>(H, DIM, DIM, ah + h * 2 * DIM,
                                        ah + h * 2 * DIM + DIM, s1, s2);
    k_stats<<<N_NODES, 256, 0, stream>>>(A16, s1, s2, rmx, rin);
    k_nn_gemm<8, 1, 1><<<N_NODES / 64, 128, 0, stream>>>(A16, Hbt, s1, s2, rmx, rin,
                                                         nullptr, Z);
  }
  k_cvt<<<cvtBlocks, 256, 0, stream>>>(Z, Xb, 0.5f, nXD);   // mean over heads

  // out-attention (56 cols, padded to 64), elu, row softmax
  k_dense<4, 0, 0><<<N_NODES / 64, 128, 0, stream>>>(Xb, W_oa, nullptr, H56, 56);
  k_cvt_t<<<dim3(N_NODES / 32, 64 / 32), tb, 0, stream>>>(H56, Hbt, 64, 1.f);
  k_s1s2<<<N_NODES, 128, 0, stream>>>(H56, 64, 56, a_oa, a_oa + 56, s1, s2);
  k_stats<<<N_NODES, 256, 0, stream>>>(A16, s1, s2, rmx, rin);
  k_nn_gemm<4, 1, 2><<<N_NODES / 64, 128, 0, stream>>>(A16, Hbt, s1, s2, rmx, rin,
                                                       nullptr, O56);
  k_softmax56<<<N_NODES, 64, 0, stream>>>(O56);

  // per-molecule readout + MLP head + sigmoid
  (void)hipMemsetAsync(mol, 0, (size_t)N_MOL * 56 * 4, stream);
  k_segsum<<<(N_NODES * 56 + 255) / 256, 256, 0, stream>>>(O56, seg, mol);
  k_head<<<N_MOL, 64, 0, stream>>>(mol, W_out, b_out, W_pr, b_pr, out);
}
